// TtMistralAttention_24867860643895
// MI455X (gfx1250) — matrix operand order repaired and checked by reading, not compiled, not run on hardware
//
#include <hip/hip_runtime.h>

typedef __attribute__((ext_vector_type(16))) __bf16 v16bf;
typedef __attribute__((ext_vector_type(8)))  float  v8f;
typedef __attribute__((ext_vector_type(4)))  int    v4i_;

#define B_   32
#define H_   32
#define KVH_ 8
#define G_   4
#define D_   128
#define SW_  4096
#define HID_ 4096
#define QKV_ 6144

// ---------------------------------------------------------------------------
// CDNA5 async global->LDS copy (ASYNCcnt domain), with safe fallback.
// Probe-discovered signature: (as1 v4i* gaddr, as3 v4i* lds, imm offset, imm cpol)
// ---------------------------------------------------------------------------
#if defined(__has_builtin)
#if __has_builtin(__builtin_amdgcn_global_load_async_to_lds_b128)
#define HAVE_ASYNC_LDS 1
#endif
#endif

#if defined(HAVE_ASYNC_LDS)
__device__ __forceinline__ void async_b128(void* lds, const void* g) {
  __builtin_amdgcn_global_load_async_to_lds_b128(
      (__attribute__((address_space(1))) v4i_*)(g),
      (__attribute__((address_space(3))) v4i_*)(lds),
      /*offset=*/0, /*cpol=*/0);
}
__device__ __forceinline__ void wait_async() {
#if __has_builtin(__builtin_amdgcn_s_wait_asynccnt)
  __builtin_amdgcn_s_wait_asynccnt(0);
#else
  asm volatile("s_wait_asynccnt 0x0" ::: "memory");
#endif
}
#else
__device__ __forceinline__ void async_b128(void* lds, const void* g) {
  *(float4*)lds = *(const float4*)g;
}
__device__ __forceinline__ void wait_async() {}
#endif

__device__ __forceinline__ v8f wmma_bf16(v16bf a, v16bf b, v8f c) {
  // D = A(16x32 bf16) x B(32x16 bf16) + C(16x16 f32)
  return __builtin_amdgcn_wmma_f32_16x16x32_bf16(false, a, false, b, (short)0, c,
                                                 false, false);
}

// ---------------------------------------------------------------------------
// C[32 x N] = A[32 x K] @ W[K x N]   (f32 row-major, bf16 WMMA inner)
// Block = 256 threads = 8 waves, owns a 32x64 output panel (2 mt x 4 nt tiles).
// Per 32-wide k-step: async-stage A slab (32x32) and W slab (32x64, pitch 68)
// into LDS with coalesced b128s, then every wave builds its WMMA fragments
// from LDS. Grid = N/64 blocks.
// A lane layout (16-bit A 16x32): m=lane&15, half=lane>>4:
//   a[0..7]  = A[m][kc + 8*half + 0..7],  a[8..15] = A[m][kc + 16 + 8*half + 0..7]
// B lane layout (32x16): b[0..7] = W[kc + 8*half + 0..7][ncol], b[8..15] = W[kc + 16 + 8*half + 0..7][ncol]
// C lane layout: c[i] = C[mt*16 + i + 8*half][ncol]
// ---------------------------------------------------------------------------
__global__ __launch_bounds__(256) void gemm32_kernel(const float* __restrict__ A,
                                                     const float* __restrict__ W,
                                                     float* __restrict__ C,
                                                     int N, int K) {
  __shared__ float sA[32 * 32];     // sA[row*32 + k]
  __shared__ float sW[32 * 68];     // sW[k*68 + n] (pitch 68 = bank-conflict pad)

  const int tid  = threadIdx.x;
  const int lane = tid & 31;
  const int wv   = tid >> 5;            // 0..7
  const int half = lane >> 4;
  const int l16  = lane & 15;
  const int mt   = wv >> 2;             // 0..1
  const int nt   = wv & 3;              // 0..3
  const int nblk = blockIdx.x;
  const int ncol = nblk * 64 + nt * 16 + l16;

  // staging assignments (coalesced 16B chunks)
  const int ar  = tid >> 3;             // A row 0..31
  const int ac4 = (tid & 7) * 4;        // A col (floats)
  const int wk  = tid >> 3;             // W k-row 0..31
  const int wc8 = (tid & 7) * 8;        // W col (floats), 8 per thread

  v8f acc = {};
  for (int kc = 0; kc < K; kc += 32) {
    async_b128(&sA[ar * 32 + ac4], A + (size_t)ar * K + kc + ac4);
    async_b128(&sW[wk * 68 + wc8],     W + (size_t)(kc + wk) * N + nblk * 64 + wc8);
    async_b128(&sW[wk * 68 + wc8 + 4], W + (size_t)(kc + wk) * N + nblk * 64 + wc8 + 4);
    wait_async();
    __syncthreads();

    v16bf a, b;
    const float* r0 = &sA[(mt * 16 + l16) * 32 + 8 * half];
#pragma unroll
    for (int i = 0; i < 8; ++i) {
      a[i]     = (__bf16)r0[i];
      a[8 + i] = (__bf16)r0[16 + i];
    }
#pragma unroll
    for (int i = 0; i < 16; ++i)
      b[i] = (__bf16)sW[((i < 8) ? (8 * half + i) : (16 + 8 * half + (i - 8))) * 68 + nt * 16 + l16];
    acc = wmma_bf16(a, b, acc);
    __syncthreads();
  }
#pragma unroll
  for (int i = 0; i < 8; ++i)
    C[(size_t)(mt * 16 + i + 8 * half) * N + ncol] = acc[i];
}

// ---------------------------------------------------------------------------
// Rotary: rows 0..1023 are q (r = h*B + b), rows 1024..1279 are k (r2 = b*8+kv)
// out_row = in_row @ rot_mat; q rows additionally scaled by D^-0.5.
// qout layout: [h][b][d] (== (KVH,G,B,D) flat); kout layout: [b][kv][d].
// 80 M-tiles x 8 N-tiles = 640 waves. rot_mat is 64 KB -> L2 resident.
// ---------------------------------------------------------------------------
__global__ __launch_bounds__(256) void rope_kernel(const float* __restrict__ xqkv,
                                                   const float* __restrict__ rot,
                                                   float* __restrict__ qout,
                                                   float* __restrict__ kout) {
  const int wave = (blockIdx.x * blockDim.x + threadIdx.x) >> 5;
  const int lane = threadIdx.x & 31;
  const int mt = wave >> 3;   // 0..79
  const int nt = wave & 7;    // 0..7
  const int half = lane >> 4;
  const int l16  = lane & 15;
  const int row  = mt * 16 + l16;
  const bool isQ = (row < H_ * B_);
  const float* src;
  if (isQ) {
    int h = row >> 5, b = row & 31;
    src = xqkv + (size_t)b * QKV_ + (size_t)h * D_;
  } else {
    int r2 = row - H_ * B_;
    int b = r2 >> 3, kv = r2 & 7;
    src = xqkv + (size_t)b * QKV_ + HID_ + (size_t)kv * D_;
  }
  v8f acc = {};
#pragma unroll
  for (int c = 0; c < 4; ++c) {
    v16bf a, bm;
    const float* r0 = src + c * 32 + 8 * half;
#pragma unroll
    for (int i = 0; i < 8; ++i) {
      a[i]     = (__bf16)r0[i];
      a[8 + i] = (__bf16)r0[16 + i];
    }
    const float* bc = rot + (size_t)(c * 32 + 8 * half) * D_ + nt * 16 + l16;
#pragma unroll
    for (int i = 0; i < 16; ++i) bm[i] = (__bf16)bc[(size_t)((i < 8) ? i : (i + 8)) * D_];
    acc = wmma_bf16(a, bm, acc);
  }
  const float qscale = 0.08838834764831845f;  // 128^-0.5
#pragma unroll
  for (int i = 0; i < 8; ++i) {
    int orow = mt * 16 + i + 8 * half;  // whole tile is same q/k segment
    if (orow < H_ * B_)
      qout[(size_t)orow * D_ + nt * 16 + l16] = acc[i] * qscale;
    else
      kout[(size_t)(orow - H_ * B_) * D_ + nt * 16 + l16] = acc[i];
  }
}

// vnew[b][kv][d] = xqkv[b][(H+KVH)*D + kv*D + d]
__global__ __launch_bounds__(256) void vcopy_kernel(const float* __restrict__ xqkv,
                                                    float* __restrict__ vout) {
  int i = blockIdx.x * blockDim.x + threadIdx.x;  // 0..32767
  int r = i >> 7, d = i & 127;
  int b = r >> 3, kv = r & 7;
  vout[i] = xqkv[(size_t)b * QKV_ + (size_t)(H_ + KVH_) * D_ + (size_t)kv * D_ + d];
}

// ---------------------------------------------------------------------------
// Flash-decode attention: one wave per (b, kv). G=4 query heads zero-padded to
// the 16-wide N dim. S^T(16pos x 16head) = K_tile x q^T via WMMA; online
// softmax over position tiles of 32; O(16head x 128d) += P x V via WMMA with
// P^T transposed through LDS and the V tile async-staged into LDS.
// Position current_pos comes from krot/vnew (inputs must not be mutated).
// ---------------------------------------------------------------------------
__global__ __launch_bounds__(32) void attn_kernel(const float* __restrict__ qrot,
                                                  const float* __restrict__ krot,
                                                  const float* __restrict__ vnew,
                                                  const float* __restrict__ cache_k,
                                                  const float* __restrict__ cache_v,
                                                  float* __restrict__ attn,
                                                  const int* __restrict__ cpos_p) {
  __shared__ float Vt[32 * D_];   // 16 KB: 32 positions x 128 dims, f32
  __shared__ float Pt[32 * 16];   // 2 KB: 32 positions x 16 heads, f32

  const int b  = blockIdx.x >> 3;
  const int kv = blockIdx.x & 7;
  const int lane = threadIdx.x;
  const int half = lane >> 4;
  const int l16  = lane & 15;
  const int cpos = cpos_p[0];
  int L = cpos + 1; if (L > SW_) L = SW_;
  const int ntiles = (L + 31) >> 5;

  // Preload q as B-matrix fragments (head = N column), zero-padded heads >= 4.
  v16bf bq[4];
  {
    const int qh = (l16 < G_) ? l16 : 0;
    const float qm = (l16 < G_) ? 1.0f : 0.0f;
    const float* qbase = qrot + ((size_t)(kv * G_ + qh) * B_ + b) * D_;
#pragma unroll
    for (int c = 0; c < 4; ++c)
#pragma unroll
      for (int i = 0; i < 16; ++i)
        bq[c][i] = (__bf16)(qbase[c * 32 + ((i < 8) ? (8 * half + i) : (16 + 8 * half + (i - 8)))] * qm);
  }

  float m_run = -INFINITY, l_run = 0.0f;
  v8f Oacc[8];
#pragma unroll
  for (int dt = 0; dt < 8; ++dt) Oacc[dt] = (v8f){};

  const size_t kvbase = (size_t)(b * KVH_ + kv) * SW_ * D_;
  const size_t newoff = (size_t)(b * KVH_ + kv) * D_;

  for (int t = 0; t < ntiles; ++t) {
    const int tbase = t * 32;

    // Async-stage V tile (32 x 128 f32) into LDS; one position row per lane.
    {
      const int p = tbase + lane;
      if (p < L) {
        const float* src = (p < cpos) ? cache_v + kvbase + (size_t)p * D_
                                      : vnew + newoff;
#pragma unroll 8
        for (int j = 0; j < 32; ++j)
          async_b128(&Vt[lane * D_ + j * 4], src + j * 4);
      } else {
        const float4 z = {0.f, 0.f, 0.f, 0.f};
        float4* dst4 = (float4*)(Vt + lane * D_);
#pragma unroll 8
        for (int j = 0; j < 32; ++j) dst4[j] = z;
      }
    }

    // Prefetch next tile's K/V rows while this tile is in flight.
    {
      const int pn = tbase + 32 + lane;
      if (pn < cpos) {
        __builtin_prefetch(cache_k + kvbase + (size_t)pn * D_, 0, 0);
        __builtin_prefetch(cache_v + kvbase + (size_t)pn * D_, 0, 0);
      }
    }

    // S^T tiles: positions tbase+pt*16+(i+8*half), head = l16.
    v8f s[2];
#pragma unroll
    for (int pt = 0; pt < 2; ++pt) {
      v8f acc = {};
      const int prow = tbase + pt * 16 + l16;  // A row = position
      const float* ksrc = (prow < cpos) ? cache_k + kvbase + (size_t)prow * D_
                        : (prow == cpos) ? krot + newoff : nullptr;
#pragma unroll
      for (int c = 0; c < 4; ++c) {
        v16bf a;
        if (ksrc) {
          const float* r0 = ksrc + c * 32 + 8 * half;
#pragma unroll
          for (int i = 0; i < 8; ++i) {
            a[i]     = (__bf16)r0[i];
            a[8 + i] = (__bf16)r0[16 + i];
          }
        } else {
#pragma unroll
          for (int i = 0; i < 16; ++i) a[i] = (__bf16)0.0f;
        }
        acc = wmma_bf16(a, bq[c], acc);
      }
      s[pt] = acc;
    }

    // Mask out-of-window positions.
#pragma unroll
    for (int pt = 0; pt < 2; ++pt)
#pragma unroll
      for (int i = 0; i < 8; ++i) {
        int pos = tbase + pt * 16 + i + 8 * half;
        if (pos >= L) s[pt][i] = -1.0e30f;
      }

    // Online softmax: per-head (column) max/sum, cross-half via shfl_xor 16.
    float tmax = -1.0e30f;
#pragma unroll
    for (int pt = 0; pt < 2; ++pt)
#pragma unroll
      for (int i = 0; i < 8; ++i) tmax = fmaxf(tmax, s[pt][i]);
    tmax = fmaxf(tmax, __shfl_xor(tmax, 16, 32));
    const float m_new = fmaxf(m_run, tmax);
    const float scale_old = __expf(m_run - m_new);
    float tsum = 0.0f;
#pragma unroll
    for (int pt = 0; pt < 2; ++pt)
#pragma unroll
      for (int i = 0; i < 8; ++i) {
        float p = __expf(s[pt][i] - m_new);
        s[pt][i] = p;
        tsum += p;
      }
    tsum += __shfl_xor(tsum, 16, 32);
    l_run = l_run * scale_old + tsum;
    m_run = m_new;

    // Write P^T tile (pos x head) to LDS for the transpose.
#pragma unroll
    for (int pt = 0; pt < 2; ++pt)
#pragma unroll
      for (int i = 0; i < 8; ++i)
        Pt[(pt * 16 + i + 8 * half) * 16 + l16] = s[pt][i];

    // V tile must have landed before B-fragment reads.
    wait_async();
    __syncthreads();

    // Rescale running O by per-head exp(m_old - m_new).
    float rs[8];
#pragma unroll
    for (int i = 0; i < 8; ++i) rs[i] = __shfl(scale_old, i + 8 * half, 32);
#pragma unroll
    for (int dt = 0; dt < 8; ++dt)
#pragma unroll
      for (int i = 0; i < 8; ++i) Oacc[dt][i] *= rs[i];

    // A = P (16 head x 32 pos) from LDS (transposed read).
    v16bf ap;
#pragma unroll
    for (int i = 0; i < 16; ++i) {
      int kpos = (i < 8) ? (8 * half + i) : (16 + 8 * half + (i - 8));
      ap[i] = (__bf16)Pt[kpos * 16 + l16];
    }
    // O(head x d) += P x V, 8 d-tiles of 16.
#pragma unroll
    for (int dt = 0; dt < 8; ++dt) {
      v16bf bv;
#pragma unroll
      for (int i = 0; i < 16; ++i)
        bv[i] = (__bf16)Vt[((i < 8) ? (8 * half + i) : (16 + 8 * half + (i - 8))) * D_ + dt * 16 + l16];
      Oacc[dt] = wmma_bf16(ap, bv, Oacc[dt]);
    }
    __syncthreads();
  }

  // Normalize by l and store the 4 real heads: attn[b][kv*4+head][d].
#pragma unroll
  for (int i = 0; i < 8; ++i) {
    const int head = i + 8 * half;
    const float lh = __shfl(l_run, head, 32);
    if (head < G_) {
      float* dst = attn + ((size_t)(b * H_) + kv * G_ + head) * D_;
#pragma unroll
      for (int dt = 0; dt < 8; ++dt) dst[dt * 16 + l16] = Oacc[dt][i] / lh;
    }
  }
}

// ---------------------------------------------------------------------------
extern "C" void kernel_launch(void* const* d_in, const int* in_sizes, int n_in,
                              void* d_out, int out_size, void* d_ws, size_t ws_size,
                              hipStream_t stream) {
  (void)in_sizes; (void)n_in; (void)out_size; (void)ws_size;
  const float* x      = (const float*)d_in[0];
  const float* wqkv   = (const float*)d_in[1];
  const float* wo     = (const float*)d_in[2];
  const float* rot    = (const float*)d_in[3];
  const float* cachek = (const float*)d_in[4];
  const float* cachev = (const float*)d_in[5];
  const int*   cpos   = (const int*)d_in[6];

  float* ws   = (float*)d_ws;
  float* xqkv = ws;                        // 32*6144   = 196608
  float* qrot = xqkv + B_ * QKV_;          // 1024*128  = 131072
  float* krot = qrot + H_ * B_ * D_;       // 256*128   =  32768
  float* vnew = krot + B_ * KVH_ * D_;     // 256*128   =  32768
  float* attn = vnew + B_ * KVH_ * D_;     // 32*4096   = 131072  (2 MB total)

  // 1) xqkv = x @ wqkv : 96 blocks x (32x64 panel)
  gemm32_kernel<<<QKV_ / 64, 256, 0, stream>>>(x, wqkv, xqkv, QKV_, HID_);
  // 2) rotary q/k: 80 x 8 tiles = 640 waves = 80 blocks
  rope_kernel<<<80, 256, 0, stream>>>(xqkv, rot, qrot, krot);
  // 3) v slice copy
  vcopy_kernel<<<128, 256, 0, stream>>>(xqkv, vnew);
  // 4) flash-decode attention: one wave per (b, kv)
  attn_kernel<<<B_ * KVH_, 32, 0, stream>>>(qrot, krot, vnew, cachek, cachev,
                                            attn, cpos);
  // 5) dense = attn @ wo : 64 blocks x (32x64 panel)
  gemm32_kernel<<<HID_ / 64, 256, 0, stream>>>(attn, wo, (float*)d_out, HID_, HID_);
}
